// PointTransformerLayer_1855425872105
// MI455X (gfx1250) — matrix operand order repaired and checked by reading, not compiled, not run on hardware
//
#include <hip/hip_runtime.h>
#include <hip/hip_bf16.h>
#include <math.h>

#define BB   4
#define NN   8192
#define CIN  64
#define COUT 64
#define KNB  16     // neighbors
#define NATT 8
#define EPSV 1e-5f

typedef __attribute__((ext_vector_type(16))) _Float16 v16h;
typedef __attribute__((ext_vector_type(8)))  float    v8f;

// ---------------------------------------------------------------------------
// CDNA5 async global->LDS copy (no VGPR round trip, tracked by ASYNCcnt).
// vdst operand = LDS byte offset (low 32 bits of a generic pointer to LDS).
// ---------------------------------------------------------------------------
__device__ __forceinline__ void async_load_lds_b128(void* lds_ptr, const void* gptr)
{
    const unsigned lds_off = (unsigned)(uintptr_t)lds_ptr;
    const unsigned long long ga = (unsigned long long)(uintptr_t)gptr;
    asm volatile("global_load_async_to_lds_b128 %0, %1, off"
                 :: "v"(lds_off), "v"(ga)
                 : "memory");
}
__device__ __forceinline__ void wait_asynccnt0()
{
    asm volatile("s_wait_asynccnt 0" ::: "memory");
}

// ---------------------------------------------------------------------------
// Kernel 1: Q/K/V projection.  out[64 x 16-tile] = W[64x64] * x[64x16] + b
// One wave per (matrix, batch, 16-point tile). 4 M-tiles x 2 K-steps of
// v_wmma_f32_16x16x32_f16, fp32 accumulation.
// ---------------------------------------------------------------------------
__global__ __launch_bounds__(256) void qkv_wmma_kernel(
    const float* __restrict__ x,
    const float* __restrict__ wq, const float* __restrict__ bq,
    const float* __restrict__ wk, const float* __restrict__ bk,
    const float* __restrict__ wv, const float* __restrict__ bv,
    float* __restrict__ qo, float* __restrict__ ko, float* __restrict__ vo)
{
    const int gwave  = (blockIdx.x * blockDim.x + threadIdx.x) >> 5;
    const int lane   = threadIdx.x & 31;
    const int ntiles = NN / 16;                       // 512

    int u = gwave;
    const int tile = u % ntiles; u /= ntiles;
    const int b    = u % BB;     u /= BB;
    const int mat  = u;                               // 0,1,2
    if (mat >= 3) return;

    const float* W   = (mat == 0) ? wq : (mat == 1) ? wk : wv;
    const float* bia = (mat == 0) ? bq : (mat == 1) ? bk : bv;
    float*       out = (mat == 0) ? qo : (mat == 1) ? ko : vo;

    const int r  = lane & 15;     // row / col-in-tile
    const int hi = lane >> 4;     // half-wave
    const int n0 = tile * 16;

    v8f acc[4];
    #pragma unroll
    for (int mt = 0; mt < 4; ++mt)
        #pragma unroll
        for (int e = 0; e < 8; ++e) acc[mt][e] = 0.0f;

    #pragma unroll
    for (int kt = 0; kt < 2; ++kt) {
        // B operand (32x16 f16): element e -> K = (e<8) ? hi*8+e : 16+hi*8+(e-8), N = r
        v16h bmat;
        #pragma unroll
        for (int e = 0; e < 16; ++e) {
            const int kk = kt * 32 + ((e < 8) ? (hi * 8 + e) : (16 + hi * 8 + (e - 8)));
            bmat[e] = (_Float16)x[((size_t)b * CIN + kk) * NN + n0 + r];
        }
        #pragma unroll
        for (int mt = 0; mt < 4; ++mt) {
            // A operand (16x32 f16): lane r = M; K per ISA 16-bit A layout
            v16h amat;
            #pragma unroll
            for (int i = 0; i < 8; ++i) {
                #pragma unroll
                for (int s = 0; s < 2; ++s) {
                    const int kk = kt * 32 +
                        ((i < 4) ? (hi * 8 + 2 * i + s)
                                 : (16 + hi * 8 + 2 * (i - 4) + s));
                    amat[2 * i + s] = (_Float16)W[(mt * 16 + r) * CIN + kk];
                }
            }
            acc[mt] = __builtin_amdgcn_wmma_f32_16x16x32_f16(
                false, amat, false, bmat, (short)0, acc[mt], false, false);
        }
    }

    // C/D layout: VGPR rr -> M = mt*16 + rr + hi*8, N = r
    #pragma unroll
    for (int mt = 0; mt < 4; ++mt) {
        #pragma unroll
        for (int rr = 0; rr < 8; ++rr) {
            const int m = mt * 16 + rr + hi * 8;
            out[((size_t)b * COUT + m) * NN + n0 + r] = acc[mt][rr] + bia[m];
        }
    }
}

// ---------------------------------------------------------------------------
// Kernel 2: KNN (top-16 by squared distance).  One thread per query point.
// Candidate tiles staged global->LDS with GLOBAL_LOAD_ASYNC_TO_LDS_B128
// (ASYNCcnt), consumed 4 candidates at a time via ds_load_b128 broadcasts.
// Branchless unrolled sorted-insert keeps stable (lowest-index-first) order.
// ---------------------------------------------------------------------------
#define CAND_TILE 2048
__global__ __launch_bounds__(256) void knn_kernel(
    const float* __restrict__ p, int* __restrict__ idx_out)
{
    __shared__ alignas(16) float sp[CAND_TILE * 3];   // interleaved xyz, 24 KB
    const int b = blockIdx.y;
    const int n = blockIdx.x * blockDim.x + threadIdx.x;
    const float* pb = p + (size_t)b * NN * 3;

    const float qx = pb[n * 3 + 0];
    const float qy = pb[n * 3 + 1];
    const float qz = pb[n * 3 + 2];

    float bd[KNB]; int bi[KNB];
    #pragma unroll
    for (int i = 0; i < KNB; ++i) { bd[i] = 3.4e38f; bi[i] = 0; }

    for (int t0 = 0; t0 < NN; t0 += CAND_TILE) {
        __syncthreads();                               // tile reuse WAR
        // async stage: 24576 B / (256 thr * 16 B) = 6 issues per thread
        for (int i = threadIdx.x * 4; i < CAND_TILE * 3; i += blockDim.x * 4)
            async_load_lds_b128(&sp[i], pb + (size_t)t0 * 3 + i);
        if (t0 + CAND_TILE < NN)
            __builtin_prefetch(&pb[(t0 + CAND_TILE) * 3], 0, 0);  // global_prefetch_b8
        wait_asynccnt0();
        __syncthreads();

        const float4* sp4 = (const float4*)sp;
        for (int g = 0; g < CAND_TILE / 4; ++g) {
            // 3 x b128 broadcast reads cover 4 interleaved xyz candidates
            const float4 A = sp4[g * 3 + 0];
            const float4 Bv = sp4[g * 3 + 1];
            const float4 C = sp4[g * 3 + 2];
            float cx[4], cy[4], cz[4];
            cx[0] = A.x;  cy[0] = A.y;  cz[0] = A.z;
            cx[1] = A.w;  cy[1] = Bv.x; cz[1] = Bv.y;
            cx[2] = Bv.z; cy[2] = Bv.w; cz[2] = C.x;
            cx[3] = C.y;  cy[3] = C.z;  cz[3] = C.w;
            #pragma unroll
            for (int u = 0; u < 4; ++u) {
                const float dx = qx - cx[u];
                const float dy = qy - cy[u];
                const float dz = qz - cz[u];
                const float d  = dx * dx + dy * dy + dz * dz;
                if (d < bd[KNB - 1]) {
                    bd[KNB - 1] = d; bi[KNB - 1] = t0 + g * 4 + u;
                    #pragma unroll
                    for (int i = KNB - 1; i > 0; --i) {
                        const bool sw = bd[i] < bd[i - 1];
                        const float td = bd[i];  const int ti = bi[i];
                        bd[i]     = sw ? bd[i - 1] : bd[i];
                        bi[i]     = sw ? bi[i - 1] : bi[i];
                        bd[i - 1] = sw ? td : bd[i - 1];
                        bi[i - 1] = sw ? ti : bi[i - 1];
                    }
                }
            }
        }
    }
    #pragma unroll
    for (int i = 0; i < KNB; ++i)
        idx_out[((size_t)b * NN + n) * KNB + i] = bi[i];
}

// ---------------------------------------------------------------------------
// Kernel 3: fused pointwise attention.  One wave per point.
// lane -> (neighbor j = lane&15, rep = lane>>4); each lane owns 32 channels.
// Gathers k/v from L2-resident arrays; all MLP params staged in 4 KB LDS.
// ---------------------------------------------------------------------------
__global__ __launch_bounds__(256) void fused_attn_kernel(
    const float* __restrict__ p, const int* __restrict__ knn,
    const float* __restrict__ q, const float* __restrict__ k,
    const float* __restrict__ v,
    const float* __restrict__ pe_w1, const float* __restrict__ pe_g1,
    const float* __restrict__ pe_b1, const float* __restrict__ pe_w2,
    const float* __restrict__ pe_b2,
    const float* __restrict__ at_g1, const float* __restrict__ at_be1,
    const float* __restrict__ at_w1, const float* __restrict__ at_g2,
    const float* __restrict__ at_be2, const float* __restrict__ at_w2,
    const float* __restrict__ at_bias,
    float* __restrict__ y)
{
    // LDS parameter layout (floats)
    __shared__ float sprm[1024];
    const float rs = rsqrtf(1.0f + EPSV);
    {
        const int t = threadIdx.x;
        if (t < 9)  sprm[t] = pe_w1[t];
        if (t < 3)  { sprm[16 + t] = pe_g1[t] * rs; sprm[20 + t] = pe_b1[t]; }
        for (int i = t; i < 192; i += blockDim.x) sprm[32 + i] = pe_w2[i];
        for (int i = t; i < 64; i += blockDim.x) {
            sprm[224 + i] = pe_b2[i];
            sprm[288 + i] = at_g1[i] * rs;
            sprm[352 + i] = at_be1[i];
            sprm[944 + i] = at_w2[i];
        }
        for (int i = t; i < 512; i += blockDim.x) sprm[416 + i] = at_w1[i];
        if (t < 8) {
            sprm[928 + t]  = at_g2[t] * rs;
            sprm[936 + t]  = at_be2[t];
            sprm[1008 + t] = at_bias[t];
        }
    }
    __syncthreads();

    const int gwave = (blockIdx.x * blockDim.x + threadIdx.x) >> 5;
    const int lane  = threadIdx.x & 31;
    const int b     = gwave / NN;
    const int n     = gwave % NN;
    const int j     = lane & 15;
    const int rep   = lane >> 4;

    const float* pb = p + (size_t)b * NN * 3;
    const int nb = knn[((size_t)b * NN + n) * KNB + j];

    const float rx = pb[nb * 3 + 0] - pb[n * 3 + 0];
    const float ry = pb[nb * 3 + 1] - pb[n * 3 + 1];
    const float rz = pb[nb * 3 + 2] - pb[n * 3 + 2];

    // positional MLP stage 1: h = relu(bn(pe_w1 @ rel))
    const float h0 = fmaxf(sprm[16] * (sprm[0]*rx + sprm[1]*ry + sprm[2]*rz) + sprm[20], 0.f);
    const float h1 = fmaxf(sprm[17] * (sprm[3]*rx + sprm[4]*ry + sprm[5]*rz) + sprm[21], 0.f);
    const float h2 = fmaxf(sprm[18] * (sprm[6]*rx + sprm[7]*ry + sprm[8]*rz) + sprm[22], 0.f);

    const float* qb = q + (size_t)b * COUT * NN;
    const float* kb = k + (size_t)b * COUT * NN;
    const float* vb = v + (size_t)b * COUT * NN;

    float sm[8];
    #pragma unroll
    for (int m = 0; m < 8; ++m) sm[m] = 0.0f;
    float nv[32];

    #pragma unroll
    for (int t = 0; t < 32; ++t) {
        const int c  = rep * 32 + t;
        const float rc = sprm[32 + c * 3 + 0] * h0 + sprm[32 + c * 3 + 1] * h1 +
                         sprm[32 + c * 3 + 2] * h2 + sprm[224 + c];
        const float kc = kb[(size_t)c * NN + nb];
        const float vc = vb[(size_t)c * NN + nb];
        const float qc = qb[(size_t)c * NN + n];
        nv[t] = vc + rc;
        const float a = fmaxf((qc - kc + rc) * sprm[288 + c] + sprm[352 + c], 0.f);
        #pragma unroll
        for (int m = 0; m < 8; ++m)
            sm[m] = fmaf(sprm[416 + m * 64 + c], a, sm[m]);
    }

    // fold the two channel-halves (lanes j and j+16)
    #pragma unroll
    for (int m = 0; m < 8; ++m) sm[m] += __shfl_xor(sm[m], 16, 32);

    // attention MLP stage 2 + logits
    float z[8];
    #pragma unroll
    for (int m = 0; m < 8; ++m)
        z[m] = fmaxf(sm[m] * sprm[928 + m] + sprm[936 + m], 0.f);
    float lg[8];
    #pragma unroll
    for (int m = 0; m < 8; ++m) {
        float s = sprm[1008 + m];
        #pragma unroll
        for (int mp = 0; mp < 8; ++mp)
            s = fmaf(sprm[944 + m * 8 + mp], z[mp], s);
        lg[m] = s;
    }

    // softmax over 16 neighbors (xor 1,2,4,8 stays inside each 16-lane group)
    float wgt[8];
    #pragma unroll
    for (int m = 0; m < 8; ++m) {
        float mx = lg[m];
        mx = fmaxf(mx, __shfl_xor(mx, 1, 32));
        mx = fmaxf(mx, __shfl_xor(mx, 2, 32));
        mx = fmaxf(mx, __shfl_xor(mx, 4, 32));
        mx = fmaxf(mx, __shfl_xor(mx, 8, 32));
        const float e = __expf(lg[m] - mx);
        float s = e;
        s += __shfl_xor(s, 1, 32);
        s += __shfl_xor(s, 2, 32);
        s += __shfl_xor(s, 4, 32);
        s += __shfl_xor(s, 8, 32);
        wgt[m] = e / s;
    }

    // weighted combine: y[c] = sum_j wgt[c>>3][j] * nv[c][j]
    float* yb = y + (size_t)b * COUT * NN;
    #pragma unroll
    for (int t = 0; t < 32; ++t) {
        const int c   = rep * 32 + t;
        const int mlo = t >> 3;                       // compile-time after unroll
        const float wm = (rep == 0) ? wgt[mlo] : wgt[mlo + 4];
        float val = wm * nv[t];
        val += __shfl_xor(val, 1, 32);
        val += __shfl_xor(val, 2, 32);
        val += __shfl_xor(val, 4, 32);
        val += __shfl_xor(val, 8, 32);
        if (j == 0) yb[(size_t)c * NN + n] = val;
    }
}

// ---------------------------------------------------------------------------
extern "C" void kernel_launch(void* const* d_in, const int* in_sizes, int n_in,
                              void* d_out, int out_size, void* d_ws, size_t ws_size,
                              hipStream_t stream)
{
    const float* p      = (const float*)d_in[0];
    const float* x      = (const float*)d_in[1];
    const float* wq     = (const float*)d_in[2];
    const float* bq     = (const float*)d_in[3];
    const float* wk     = (const float*)d_in[4];
    const float* bk     = (const float*)d_in[5];
    const float* wv     = (const float*)d_in[6];
    const float* bv     = (const float*)d_in[7];
    const float* pe_w1  = (const float*)d_in[8];
    const float* pe_g1  = (const float*)d_in[9];
    const float* pe_b1  = (const float*)d_in[10];
    const float* pe_w2  = (const float*)d_in[11];
    const float* pe_b2  = (const float*)d_in[12];
    const float* at_g1  = (const float*)d_in[13];
    const float* at_be1 = (const float*)d_in[14];
    const float* at_w1  = (const float*)d_in[15];
    const float* at_g2  = (const float*)d_in[16];
    const float* at_be2 = (const float*)d_in[17];
    const float* at_w2  = (const float*)d_in[18];
    const float* at_bias= (const float*)d_in[19];

    float* y = (float*)d_out;

    const size_t qkv_elems = (size_t)BB * COUT * NN;      // 2,097,152 each
    float* q  = (float*)d_ws;
    float* k  = q + qkv_elems;
    float* v  = k + qkv_elems;
    int*  idx = (int*)(v + qkv_elems);                     // B*N*16 ints

    // Kernel 1: 3 mats * 4 batches * 512 tiles = 6144 waves -> 768 blocks x 256
    qkv_wmma_kernel<<<768, 256, 0, stream>>>(x, wq, bq, wk, bk, wv, bv, q, k, v);

    // Kernel 2: one thread per query; grid (8192/256, B)
    knn_kernel<<<dim3(NN / 256, BB), 256, 0, stream>>>(p, idx);

    // Kernel 3: one wave per point: B*N waves -> 4096 blocks x 256
    fused_attn_kernel<<<(BB * NN) / 8, 256, 0, stream>>>(
        p, idx, q, k, v,
        pe_w1, pe_g1, pe_b1, pe_w2, pe_b2,
        at_g1, at_be1, at_w1, at_g2, at_be2, at_w2, at_bias, y);
}
